// VecKM_71184787964234
// MI455X (gfx1250) — matrix operand order repaired and checked by reading, not compiled, not run on hardware
//
#include <hip/hip_runtime.h>
#include <math.h>

typedef __attribute__((ext_vector_type(16))) _Float16 v16h;
typedef __attribute__((ext_vector_type(8)))  float    v8f;

#define NPTS   8192
#define DDIM   384
#define RADIUS2 0.04f

// ---------------------------------------------------------------------------
// Phase 1: pA = pts @ A ; cT[j*N+i] = (f16)cos(pA[i][j]), sT likewise sin.
// Stored TRANSPOSED (D x N) so phase-2 WMMA B-fragments are contiguous loads.
// ---------------------------------------------------------------------------
__global__ __launch_bounds__(256)
void veckm_phase1(const float* __restrict__ pts, const float* __restrict__ A,
                  _Float16* __restrict__ cT, _Float16* __restrict__ sT) {
    const int idx = blockIdx.x * 256 + threadIdx.x;   // covers N*D exactly
    const int i = idx & (NPTS - 1);                   // point index (fast-varying)
    const int j = idx >> 13;                          // feature index (uniform per block)
    const float px = pts[i * 3 + 0];
    const float py = pts[i * 3 + 1];
    const float pz = pts[i * 3 + 2];
    const float pa = px * A[j] + py * A[DDIM + j] + pz * A[2 * DDIM + j];
    float s, c;
    sincosf(pa, &s, &c);
    cT[(size_t)j * NPTS + i] = (_Float16)c;
    sT[(size_t)j * NPTS + i] = (_Float16)s;
}

// ---------------------------------------------------------------------------
// Phase 2: block = 32 output rows x 384 cols. 8 waves x 3 col-tiles each,
// 2 row-tiles -> every B fragment feeds 2 WMMAs (12 wmma : 12 b128-loads).
// Per 32-wide k-step: build two 16x32 J tiles (f16 0/1) in double-buffered
// LDS in the WMMA A-fragment layout (ONE barrier per step), then WMMA.
// Epilogue: row-norm reduction + e^{-i pA} rotation + scale, write complex.
// ---------------------------------------------------------------------------
__global__ __launch_bounds__(256)
void veckm_phase2(const float* __restrict__ pts, const float* __restrict__ A,
                  const _Float16* __restrict__ cT, const _Float16* __restrict__ sT,
                  float2* __restrict__ out) {
    __shared__ float kxs[NPTS];                 // all points' y-coord (32 KB)
    __shared__ float kys[NPTS];                 // all points' z-coord (32 KB)
    // Double-buffered J: [parity][row-tile][lane][8 uints] = 4 KB
    __shared__ __align__(32) unsigned int Jf[2 * 2 * 256];
    __shared__ float rowpts[32][3];
    __shared__ float rownorm[32];

    const int tid  = threadIdx.x;
    const int lane = tid & 31;
    const int wave = tid >> 5;                  // 0..7
    const int r0   = blockIdx.x * 32;           // first output row of this block

    // Cache every point's (y,z) in LDS once (fits easily in 320KB WGP LDS).
    for (int i = tid; i < NPTS; i += 256) {
        kxs[i] = pts[i * 3 + 1];
        kys[i] = pts[i * 3 + 2];
    }
    if (tid < 32) {
        rowpts[tid][0] = pts[(r0 + tid) * 3 + 0];
        rowpts[tid][1] = pts[(r0 + tid) * 3 + 1];
        rowpts[tid][2] = pts[(r0 + tid) * 3 + 2];
        rownorm[tid] = 0.0f;
    }
    __syncthreads();

    const int hi = (lane >= 16) ? 1 : 0;
    const int n  = lane & 15;                   // tile-local row (A) / col (B/C)

    // J-build role: lane = A-fragment lane, wave index = half-pair p of the
    // 16 halves; builds one entry-pair for each of the two row tiles.
    const int h1   = 2 * wave;                                   // 0,2,..,14
    const int koff = hi * 16 + h1;                               // K of first elem
    const float rx0 = rowpts[n][1],      ry0 = rowpts[n][2];
    const float rx1 = rowpts[n + 16][1], ry1 = rowpts[n + 16][2];

    // B-fragment base offsets into cT/sT (D x N, f16): lane holds column
    // (ct*16 + n), halves are K = kb + hi*16 + 0..15 -> contiguous 32 bytes.
    const int ct0 = wave * 3;
    size_t boff[3];
#pragma unroll
    for (int t = 0; t < 3; ++t)
        boff[t] = (size_t)((ct0 + t) * 16 + n) * NPTS + hi * 16;

    v8f accRe[2][3];
    v8f accIm[2][3];
#pragma unroll
    for (int rt = 0; rt < 2; ++rt)
#pragma unroll
        for (int t = 0; t < 3; ++t) {
            accRe[rt][t] = (v8f){0,0,0,0,0,0,0,0};
            accIm[rt][t] = (v8f){0,0,0,0,0,0,0,0};
        }

    for (int kb = 0; kb < NPTS; kb += 32) {
        unsigned int* Jb = &Jf[((kb >> 5) & 1) * 512];   // this step's buffer
        {
            const int k1 = kb + koff;
            const int k2 = k1 + 1;
            const float ax = kxs[k1], ay = kys[k1];
            const float bx = kxs[k2], by = kys[k2];
            float dx, dy;
            dx = rx0 - ax; dy = ry0 - ay;
            const unsigned int p00 = (dx * dx + dy * dy < RADIUS2) ? 0x3C00u : 0u;
            dx = rx0 - bx; dy = ry0 - by;
            const unsigned int p01 = (dx * dx + dy * dy < RADIUS2) ? 0x3C00u : 0u;
            dx = rx1 - ax; dy = ry1 - ay;
            const unsigned int p10 = (dx * dx + dy * dy < RADIUS2) ? 0x3C00u : 0u;
            dx = rx1 - bx; dy = ry1 - by;
            const unsigned int p11 = (dx * dx + dy * dy < RADIUS2) ? 0x3C00u : 0u;
            Jb[lane * 8 + wave]       = p00 | (p01 << 16);
            Jb[256 + lane * 8 + wave] = p10 | (p11 << 16);
        }
        // ONE barrier per step: previous-parity buffer was fully consumed
        // before any wave signalled the previous step's barrier (the wmma's
        // s_wait_dscnt precedes the signal), so writing it now is safe.
        __syncthreads();

        const v16h a0 = *reinterpret_cast<const v16h*>(&Jb[lane * 8]);
        const v16h a1 = *reinterpret_cast<const v16h*>(&Jb[256 + lane * 8]);
#pragma unroll
        for (int t = 0; t < 3; ++t) {
            const v16h bc = *reinterpret_cast<const v16h*>(cT + boff[t] + kb);
            const v16h bs = *reinterpret_cast<const v16h*>(sT + boff[t] + kb);
            accRe[0][t] = __builtin_amdgcn_wmma_f32_16x16x32_f16(
                false, a0, false, bc, (short)0, accRe[0][t], false, false);
            accRe[1][t] = __builtin_amdgcn_wmma_f32_16x16x32_f16(
                false, a1, false, bc, (short)0, accRe[1][t], false, false);
            accIm[0][t] = __builtin_amdgcn_wmma_f32_16x16x32_f16(
                false, a0, false, bs, (short)0, accIm[0][t], false, false);
            accIm[1][t] = __builtin_amdgcn_wmma_f32_16x16x32_f16(
                false, a1, false, bs, (short)0, accIm[1][t], false, false);
        }
    }

    // ---- row-norm reduction (rotation by unit complex preserves modulus) ----
#pragma unroll
    for (int rt = 0; rt < 2; ++rt)
#pragma unroll
        for (int v = 0; v < 8; ++v) {
            float acc = 0.0f;
#pragma unroll
            for (int t = 0; t < 3; ++t)
                acc += accRe[rt][t][v] * accRe[rt][t][v]
                     + accIm[rt][t][v] * accIm[rt][t][v];
            atomicAdd(&rownorm[rt * 16 + v + 8 * hi], acc);   // ds_add_f32
        }
    __syncthreads();

    // ---- rotate, normalize, write interleaved complex output ----
    const float sd = sqrtf((float)DDIM);
#pragma unroll
    for (int rt = 0; rt < 2; ++rt)
#pragma unroll
        for (int v = 0; v < 8; ++v) {
            const int m = rt * 16 + v + 8 * hi;
            const float scale = sd * rsqrtf(rownorm[m]);
            const float px = rowpts[m][0];
            const float py = rowpts[m][1];
            const float pz = rowpts[m][2];
#pragma unroll
            for (int t = 0; t < 3; ++t) {
                const int col = (ct0 + t) * 16 + n;
                const float pa = px * A[col] + py * A[DDIM + col] + pz * A[2 * DDIM + col];
                float sv, cv;
                sincosf(pa, &sv, &cv);
                const float gre = accRe[rt][t][v];
                const float gim = accIm[rt][t][v];
                out[(size_t)(r0 + m) * DDIM + col] =
                    make_float2((gre * cv + gim * sv) * scale,
                                (gim * cv - gre * sv) * scale);
            }
        }
}

// ---------------------------------------------------------------------------
extern "C" void kernel_launch(void* const* d_in, const int* in_sizes, int n_in,
                              void* d_out, int out_size, void* d_ws, size_t ws_size,
                              hipStream_t stream) {
    const float* pts = (const float*)d_in[0];   // (N,3) f32
    const float* A   = (const float*)d_in[1];   // (3,D) f32

    _Float16* cT = (_Float16*)d_ws;                       // D x N f16 (6 MB)
    _Float16* sT = cT + (size_t)DDIM * NPTS;              // D x N f16 (6 MB)
    float2* out  = (float2*)d_out;                        // (N,D) complex64

    veckm_phase1<<<(NPTS * DDIM) / 256, 256, 0, stream>>>(pts, A, cT, sT);
    veckm_phase2<<<NPTS / 32, 256, 0, stream>>>(pts, A, cT, sT, out);
}
